// ItemBlock_12309376270654
// MI455X (gfx1250) — matrix operand order repaired and checked by reading, not compiled, not run on hardware
//
#include <hip/hip_runtime.h>

#define B_    4096
#define N_    256
#define T_    65536
#define F_    16
#define K_    64
#define DMODEL 128
#define DFF    256
#define DIN    19
#define KPAD   32
#define CLIP_  5.0f

typedef _Float16 f16;
typedef __attribute__((ext_vector_type(16))) _Float16 v16h;
typedef __attribute__((ext_vector_type(8)))  float    v8f;
typedef __attribute__((ext_vector_type(4)))  unsigned int v4u;

union V16H { v16h v; v4u q[2]; };

// ---- WMMA fragment loaders -------------------------------------------------
// A (16x32 f16, MxK): lanes 0-15 = M0..15 / K{0..7,16..23}; lanes 16-31 = K{8..15,24..31}
__device__ __forceinline__ v16h load_A_lds(const f16* base, int strideH, int mtile,
                                           int k0, int lane) {
    int m  = mtile * 16 + (lane & 15);
    int kh = (lane >> 4) * 8;
    const f16* p = base + m * strideH + k0 + kh;
    V16H r;
    r.q[0] = *(const v4u*)p;          // K = k0+kh   .. k0+kh+7
    r.q[1] = *(const v4u*)(p + 16);   // K = k0+kh+16.. k0+kh+23
    return r.v;
}

// B (32x16 f16, KxN) from row-major W[n][k] in global (W itself, since B = W^T):
// lanes 0-15 = N0..15 / K{0..7,16..23} ; lanes 16-31 = K{8..15,24..31}
__device__ __forceinline__ v16h load_B_glb(const f16* base, int strideH, int ntile,
                                           int k0, int lane) {
    int n  = ntile * 16 + (lane & 15);
    int kb = k0 + (lane >> 4) * 8;
    const f16* p = base + (size_t)n * strideH + kb;
    V16H r;
    r.q[0] = *(const v4u*)p;
    r.q[1] = *(const v4u*)(p + 16);
    return r.v;
}

__device__ __forceinline__ v8f wmma32(v16h a, v16h b, v8f c) {
    return __builtin_amdgcn_wmma_f32_16x16x32_f16(false, a, false, b,
                                                  (short)0, c, false, false);
}

// ---- weight f32 -> f16 conversion (padded W_emb) ---------------------------
__global__ void convert_weights_kernel(const float* __restrict__ W_emb,
                                       const float* __restrict__ W1,
                                       const float* __restrict__ W2,
                                       f16* __restrict__ ws) {
    int i = blockIdx.x * 256 + threadIdx.x;
    const int SZ_EMB = DMODEL * KPAD;              // 4096
    const int SZ_W1  = DFF * DMODEL;               // 32768
    const int SZ_W2  = DMODEL * DFF;               // 32768
    if (i < SZ_EMB) {
        int r = i >> 5, c = i & 31;
        ws[i] = (c < DIN) ? (f16)W_emb[r * DIN + c] : (f16)0.0f;
    } else if (i < SZ_EMB + SZ_W1) {
        ws[i] = (f16)W1[i - SZ_EMB];
    } else if (i < SZ_EMB + SZ_W1 + SZ_W2) {
        ws[i] = (f16)W2[i - SZ_EMB - SZ_W1];
    }
}

// ---- main fused kernel: one workgroup (8 wave32) per batch -----------------
__global__ __launch_bounds__(256) void item_block_kernel(
    const float* __restrict__ x,          // (T,16)
    const int*   __restrict__ indices,    // (B,256)
    const float* __restrict__ origin,     // (B,2)
    const float* __restrict__ direction,  // (B,2)
    const float* __restrict__ norm_count, // (1,)
    const float* __restrict__ norm_mean,  // (19,)
    const float* __restrict__ norm_sqsum, // (19,)
    const float* __restrict__ b_emb,      // (128,)
    const float* __restrict__ ln1_g, const float* __restrict__ ln1_b,
    const float* __restrict__ b1,         // (256,)
    const float* __restrict__ b2,         // (128,)
    const float* __restrict__ ln2_g, const float* __restrict__ ln2_b,
    const f16*   __restrict__ WembH,      // (128,32) f16
    const f16*   __restrict__ W1H,        // (256,128) f16
    const f16*   __restrict__ W2H,        // (128,256) f16
    float*       __restrict__ out)        // [B*64*128 | B*64*2 | B*64]
{
    __shared__ __align__(16) float skey[N_];
    __shared__ __align__(16) float smask[K_];
    __shared__ __align__(16) float smu[K_];
    __shared__ __align__(16) float srs[K_];
    __shared__ __align__(16) f16   Abuf[K_ * KPAD];     //  4 KB
    __shared__ __align__(16) f16   Hh[K_ * DMODEL];     // 16 KB
    __shared__ __align__(16) f16   H2h[K_ * DFF];       // 32 KB
    __shared__ __align__(16) float Hf[K_ * DMODEL];     // 32 KB

    const int b    = blockIdx.x;
    const int t    = threadIdx.x;          // 0..255
    const int lane = t & 31;
    const int wv   = t >> 5;               // 0..7

    const size_t OUT1 = (size_t)B_ * K_ * DMODEL;
    const size_t OUT2 = OUT1 + (size_t)B_ * K_ * 2;

    // ---- phase 0: per-point features + sort key ----
    const float ox = origin[2 * b],    oy = origin[2 * b + 1];
    const float dx = direction[2 * b], dy = direction[2 * b + 1];
    const float rinv = rsqrtf(dx * dx + dy * dy + 1e-16f);
    const float cth = dx * rinv, sth = -dy * rinv;   // cos/sin of -atan2(dy,dx)

    const int gi = indices[(size_t)b * N_ + t];
    const float px = x[(size_t)gi * F_ + 0];
    const float py = x[(size_t)gi * F_ + 1];
    const float present = x[(size_t)gi * F_ + 2];
    const float d0 = px - ox, d1 = py - oy;
    const float rx = cth * d0 - sth * d1;
    const float ry = sth * d0 + cth * d1;
    const float dist = sqrtf(rx * rx + ry * ry);
    const float key  = dist + ((present == 0.0f) ? 1e8f : 0.0f);
    skey[t] = key;
    __syncthreads();

    // ---- phase 1: exact stable rank (ascending key, tie -> lower index) ----
    int rank = 0;
    #pragma unroll 8
    for (int j = 0; j < N_; ++j) {
        float kj = skey[j];
        rank += (kj < key) || (kj == key && j < t);
    }
    if (rank < K_) {
        const float ncount = norm_count[0];
        const float dq  = sqrtf(dist);
        const float di  = 1.0f / (dist + 1e-8f);
        float xc[DIN];
        #pragma unroll
        for (int f = 0; f < F_; ++f) xc[f] = x[(size_t)gi * F_ + f];
        xc[16] = rx * di; xc[17] = ry * di; xc[18] = dq;
        #pragma unroll
        for (int f = 0; f < DIN; ++f) {
            float sd = sqrtf(norm_sqsum[f] / (ncount - 1.0f));
            if (sd == 0.0f) sd = 1.0f;
            float v = (xc[f] - norm_mean[f]) / sd;
            v = fminf(CLIP_, fmaxf(-CLIP_, v));
            Abuf[rank * KPAD + f] = (f16)v;
        }
        #pragma unroll
        for (int f = DIN; f < KPAD; ++f) Abuf[rank * KPAD + f] = (f16)0.0f;
        const float mk = (present == 0.0f) ? 1.0f : 0.0f;
        smask[rank] = mk;
        const size_t ro = OUT1 + ((size_t)b * K_ + rank) * 2;
        out[ro]     = rx;
        out[ro + 1] = ry;
        out[OUT2 + (size_t)b * K_ + rank] = mk;
    }
    __syncthreads();

    // ---- GEMM1: h = relu(xn @ Wemb^T + b_emb)  (64x32 @ 32x128) ----
    #pragma unroll
    for (int i = 0; i < 4; ++i) {
        const int tt = wv * 4 + i;
        const int mtile = tt >> 3, ntile = tt & 7;
        v16h a  = load_A_lds(Abuf, KPAD, mtile, 0, lane);
        v16h bf = load_B_glb(WembH, KPAD, ntile, 0, lane);
        v8f acc = {};
        acc = wmma32(a, bf, acc);
        const int col  = ntile * 16 + (lane & 15);
        const float bias = b_emb[col];
        #pragma unroll
        for (int v = 0; v < 8; ++v) {
            const int row = mtile * 16 + ((lane >> 4) ? v + 8 : v);
            Hf[row * DMODEL + col] = fmaxf(acc[v] + bias, 0.0f);
        }
    }
    __syncthreads();

    // ---- LN1 -> Hf (f32 residual source) and Hh (f16 A of GEMM2) ----
    if (t < K_) {
        float mu = 0.0f;
        for (int j = 0; j < DMODEL; ++j) mu += Hf[t * DMODEL + j];
        mu *= (1.0f / DMODEL);
        float var = 0.0f;
        for (int j = 0; j < DMODEL; ++j) {
            float d = Hf[t * DMODEL + j] - mu; var += d * d;
        }
        var *= (1.0f / DMODEL);
        const float rs = rsqrtf(var + 1e-5f);
        for (int j = 0; j < DMODEL; ++j) {
            float v = (Hf[t * DMODEL + j] - mu) * rs * ln1_g[j] + ln1_b[j];
            Hf[t * DMODEL + j] = v;
            Hh[t * DMODEL + j] = (f16)v;
        }
    }
    __syncthreads();

    // ---- GEMM2: h2 = relu(h @ W1^T + b1)  (64x128 @ 128x256) ----
    #pragma unroll
    for (int i = 0; i < 8; ++i) {
        const int tt = wv * 8 + i;
        const int mtile = tt >> 4, ntile = tt & 15;
        v8f acc = {};
        #pragma unroll
        for (int k0 = 0; k0 < DMODEL; k0 += 32) {
            v16h a  = load_A_lds(Hh, DMODEL, mtile, k0, lane);
            v16h bf = load_B_glb(W1H, DMODEL, ntile, k0, lane);
            acc = wmma32(a, bf, acc);
        }
        const int col  = ntile * 16 + (lane & 15);
        const float bias = b1[col];
        #pragma unroll
        for (int v = 0; v < 8; ++v) {
            const int row = mtile * 16 + ((lane >> 4) ? v + 8 : v);
            H2h[row * DFF + col] = (f16)fmaxf(acc[v] + bias, 0.0f);
        }
    }
    __syncthreads();

    // ---- GEMM3: resid = h + relu(h2 @ W2^T + b2)  (64x256 @ 256x128) ----
    #pragma unroll
    for (int i = 0; i < 4; ++i) {
        const int tt = wv * 4 + i;
        const int mtile = tt >> 3, ntile = tt & 7;
        v8f acc = {};
        #pragma unroll
        for (int k0 = 0; k0 < DFF; k0 += 32) {
            v16h a  = load_A_lds(H2h, DFF, mtile, k0, lane);
            v16h bf = load_B_glb(W2H, DFF, ntile, k0, lane);
            acc = wmma32(a, bf, acc);
        }
        const int col  = ntile * 16 + (lane & 15);
        const float bias = b2[col];
        #pragma unroll
        for (int v = 0; v < 8; ++v) {
            const int row = mtile * 16 + ((lane >> 4) ? v + 8 : v);
            Hf[row * DMODEL + col] += fmaxf(acc[v] + bias, 0.0f);  // in-place residual
        }
    }
    __syncthreads();

    // ---- LN2 stats ----
    if (t < K_) {
        float mu = 0.0f;
        for (int j = 0; j < DMODEL; ++j) mu += Hf[t * DMODEL + j];
        mu *= (1.0f / DMODEL);
        float var = 0.0f;
        for (int j = 0; j < DMODEL; ++j) {
            float d = Hf[t * DMODEL + j] - mu; var += d * d;
        }
        var *= (1.0f / DMODEL);
        smu[t] = mu;
        srs[t] = rsqrtf(var + 1e-5f);
    }
    __syncthreads();

    // ---- LN2 apply + mask + coalesced store ----
    for (int e = t; e < K_ * DMODEL; e += 256) {
        const int row = e >> 7, col = e & (DMODEL - 1);
        const float keep = 1.0f - smask[row];
        const float v = ((Hf[e] - smu[row]) * srs[row] * ln2_g[col] + ln2_b[col]) * keep;
        out[((size_t)b * K_ + row) * DMODEL + col] = v;
    }
}

extern "C" void kernel_launch(void* const* d_in, const int* in_sizes, int n_in,
                              void* d_out, int out_size, void* d_ws, size_t ws_size,
                              hipStream_t stream) {
    (void)in_sizes; (void)n_in; (void)out_size; (void)ws_size;
    const float* x          = (const float*)d_in[0];
    const int*   indices    = (const int*)  d_in[1];
    const float* origin     = (const float*)d_in[2];
    const float* direction  = (const float*)d_in[3];
    const float* norm_count = (const float*)d_in[4];
    const float* norm_mean  = (const float*)d_in[5];
    const float* norm_sqsum = (const float*)d_in[6];
    const float* W_emb      = (const float*)d_in[7];
    const float* b_emb      = (const float*)d_in[8];
    const float* ln1_g      = (const float*)d_in[9];
    const float* ln1_b      = (const float*)d_in[10];
    const float* W1         = (const float*)d_in[11];
    const float* b1         = (const float*)d_in[12];
    const float* W2         = (const float*)d_in[13];
    const float* b2         = (const float*)d_in[14];
    const float* ln2_g      = (const float*)d_in[15];
    const float* ln2_b      = (const float*)d_in[16];
    float* out = (float*)d_out;

    f16* wsH   = (f16*)d_ws;
    f16* WembH = wsH;                       // 128*32
    f16* W1H   = wsH + 128 * 32;            // 256*128
    f16* W2H   = wsH + 128 * 32 + 256*128;  // 128*256

    const int convN = 128*32 + 256*128 + 128*256;   // 69632
    convert_weights_kernel<<<(convN + 255) / 256, 256, 0, stream>>>(W_emb, W1, W2, wsH);

    item_block_kernel<<<B_, 256, 0, stream>>>(
        x, indices, origin, direction, norm_count, norm_mean, norm_sqsum,
        b_emb, ln1_g, ln1_b, b1, b2, ln2_g, ln2_b,
        WembH, W1H, W2H, out);
}
